// Table2Vec_1623497638531
// MI455X (gfx1250) — matrix operand order repaired and checked by reading, not compiled, not run on hardware
//
#include <hip/hip_runtime.h>
#include <hip/hip_bf16.h>
#include <math.h>

typedef __attribute__((ext_vector_type(16))) _Float16 v16h;
typedef __attribute__((ext_vector_type(8)))  _Float16 v8h;
typedef __attribute__((ext_vector_type(8)))  float    v8f;

// Problem constants
#define VV 1000000
#define EE 256
#define BB 128
#define HH 50
#define WW 9
#define LL 20
// conv output 48 x 8, 128 channels; pooled 24 x 4 -> 12288 features

// ---------------------------------------------------------------------------
// 1) Embedding gather + mean over L tokens.  One block per (b,h,w) cell,
//    lane = embedding channel (256 threads = one full 1KB row).  This is the
//    HBM-bound stage (~1.18 GB of random row reads -> ~50us at 23.3 TB/s).
//    Index loads are block-uniform -> scalarizable; all 20 row loads are
//    issued before the serial FP-add chain consumes them.
// ---------------------------------------------------------------------------
__global__ __launch_bounds__(256) void gather_mean_kernel(
    const int* __restrict__ idx, const float* __restrict__ table,
    _Float16* __restrict__ emb) {
  const int cell = blockIdx.x;
  const int tid  = threadIdx.x;
  const int* __restrict__ ip = idx + cell * LL;   // block-uniform addresses
  float v[LL];
  #pragma unroll
  for (int l = 0; l < LL; ++l) {
    v[l] = table[(size_t)ip[l] * EE + tid];
  }
  float acc = 0.0f;
  #pragma unroll
  for (int l = 0; l < LL; ++l) acc += v[l];
  emb[(size_t)cell * EE + tid] = (_Float16)(acc * (1.0f / (float)LL));
}

// ---------------------------------------------------------------------------
// 2a) Repack conv_w [oc][e][kh][kw] (OIHW f32) -> f16 [oc][k] with
//     k = (kh*2 + kw)*256 + e, so a lane's B-fragment is 32B contiguous.
// ---------------------------------------------------------------------------
__global__ void convw_transform(const float* __restrict__ w,
                                _Float16* __restrict__ out) {
  int t = blockIdx.x * blockDim.x + threadIdx.x;
  if (t >= 128 * 1536) return;
  int k  = t % 1536;
  int oc = t / 1536;
  int e   = k & 255;
  int pos = k >> 8;       // 0..5
  int kh  = pos >> 1;     // 0..2
  int kw  = pos & 1;      // 0..1
  out[t] = (_Float16)w[((oc * 256 + e) * 3 + kh) * 2 + kw];
}

// 2b) fc1_w f32 [128][12288] -> f16 same layout (row = output neuron).
__global__ void fc1w_transform(const float* __restrict__ w,
                               _Float16* __restrict__ out) {
  int t = blockIdx.x * blockDim.x + threadIdx.x;
  if (t >= 128 * 12288) return;
  out[t] = (_Float16)w[t];
}

// ---------------------------------------------------------------------------
// 3) Conv 3x2 as implicit GEMM with v_wmma_f32_16x16x32_f16, with bias +
//    2x2 maxpool + relu fused at the tail (the block tile is exactly one
//    row of pooling windows).
//
//    Block = 16 output pixels (2 oh-rows x 8 ow) x 128 oc (8 waves x 16).
//    The emb halo (4 rows x 9 w x 256 e f16 = 18KB, fully contiguous in
//    global memory) is staged into LDS with GLOBAL_LOAD_ASYNC_TO_LDS_B128
//    (ASYNCcnt path -- no VGPR round trip, no per-element wait), then each
//    wave runs 48 K-steps of WMMA.
//
//    A layout (16-bit 16x32): lane<16  -> row M=lane,    K 0..15
//                             lane>=16 -> row M=lane-16, K 16..31
//    B layout (16-bit 32x16): lane<16  -> col N=lane,    K 0..15
//                             lane>=16 -> col N=lane-16, K 16..31
//    C layout (f32 16x16):    VGPR r -> M = r + 8*(lane>=16), N = lane&15
//    Pool: vertical max = shfl_xor(16) (oh pair lives in lane / lane+16),
//          horizontal max = acc[2a] vs acc[2a+1] within the lane.
// ---------------------------------------------------------------------------
__global__ __launch_bounds__(256) void conv_wmma_kernel(
    const _Float16* __restrict__ emb, const _Float16* __restrict__ w2,
    const float* __restrict__ conv_b, _Float16* __restrict__ pooled) {
  __shared__ alignas(16) _Float16 tile[4 * 9 * 256];   // 18 KB halo

  const int t   = blockIdx.x;       // 0..3071
  const int b   = t / 24;
  const int s   = t % 24;           // pooled row ph == s
  const int oh0 = s * 2;            // conv rows oh0, oh0+1 (+2 halo)
  const int tid = threadIdx.x;

  // Async-stage 18KB: global offset is linear in chunk index i
  // (base + i*8 f16), LDS dest is tile + i*16 bytes.
  {
    const _Float16* src = emb + (size_t)(b * HH + oh0) * (WW * EE);
    const uint32_t lds_base = (uint32_t)(uintptr_t)tile;
    for (int i = tid; i < 1152; i += 256) {
      uint64_t ga = (uint64_t)(uintptr_t)(src + (size_t)i * 8);
      uint32_t la = lds_base + (uint32_t)i * 16u;
      asm volatile("global_load_async_to_lds_b128 %0, %1, off"
                   :: "v"(la), "v"(ga) : "memory");
    }
    asm volatile("s_wait_asynccnt 0x0" ::: "memory");
  }
  __syncthreads();

  const int wave = tid >> 5;
  const int lane = tid & 31;
  const int lh   = lane & 15;
  const int hi   = lane >> 4;       // 0 or 1
  const int n0   = wave * 16;

  const int m   = lh;               // this lane's A row (output pixel in tile)
  const int r_m = m >> 3;           // local oh (0/1)
  const int w_m = m & 7;            // ow
  const int oc  = n0 + lh;
  const _Float16* wrow = w2 + (size_t)oc * 1536;

  v8f acc = {};
  for (int kc = 0; kc < 48; ++kc) {
    const int k   = kc * 32;
    const int pos = k >> 8;         // (kh,kw) slice
    const int e0  = k & 255;
    const int kh  = pos >> 1;
    const int kw  = pos & 1;
    const int rr  = r_m + kh;       // 0..3
    const int ww  = w_m + kw;       // 0..8

    const _Float16* abase = tile + (((rr * 9 + ww) << 8) + e0 + (hi << 4));
    v8h alo = *reinterpret_cast<const v8h*>(abase);        // K {0..7} / {16..23}
    v8h ahi = *reinterpret_cast<const v8h*>(abase + 8);    // K {8..15} / {24..31}
    v16h a;
    #pragma unroll
    for (int i = 0; i < 8; ++i) { a[i] = alo[i]; a[8 + i] = ahi[i]; }

    v16h bf = *reinterpret_cast<const v16h*>(wrow + k + (hi << 4));

    acc = __builtin_amdgcn_wmma_f32_16x16x32_f16(
        false, a, false, bf, (short)0, acc, false, false);
  }

  // Fused bias + 2x2 maxpool + relu + f16 convert.
  const float bias = conv_b[oc];
  float mx[8];
  #pragma unroll
  for (int r = 0; r < 8; ++r) {
    float v0 = acc[r] + bias;                 // conv(oh = oh0 + hi, ow = r)
    float v1 = __shfl_xor(v0, 16, 32);        // partner row (other hi)
    mx[r] = fmaxf(v0, v1);                    // vertical max over oh pair
  }
  if (hi == 0) {
    #pragma unroll
    for (int a = 0; a < 4; ++a) {
      float p = fmaxf(fmaxf(mx[2 * a], mx[2 * a + 1]), 0.0f);
      // NCHW flatten order: j = oc*96 + ph*4 + pw
      pooled[(size_t)b * 12288 + oc * 96 + s * 4 + a] = (_Float16)p;
    }
  }
}

// ---------------------------------------------------------------------------
// 4) FC1: [128 x 12288] x [12288 x 128] with WMMA f16. 8 blocks (M tiles)
//    x 8 waves (N tiles), 384 K-steps. B matrix (3MB) is L2 resident.
// ---------------------------------------------------------------------------
__global__ __launch_bounds__(256) void fc1_wmma_kernel(
    const _Float16* __restrict__ pooled, const _Float16* __restrict__ w,
    const float* __restrict__ bias, float* __restrict__ out1) {
  const int m0   = blockIdx.x * 16;
  const int tid  = threadIdx.x;
  const int wave = tid >> 5;
  const int lane = tid & 31;
  const int lh   = lane & 15;
  const int hi   = lane >> 4;
  const int n0   = wave * 16;

  const _Float16* arow = pooled + (size_t)(m0 + lh) * 12288;
  const _Float16* brow = w      + (size_t)(n0 + lh) * 12288;

  v8f acc = {};
  for (int k = 0; k < 12288; k += 32) {
    const _Float16* ab = arow + k + (hi << 4);
    v8h alo = *reinterpret_cast<const v8h*>(ab);
    v8h ahi = *reinterpret_cast<const v8h*>(ab + 8);
    v16h a;
    #pragma unroll
    for (int i = 0; i < 8; ++i) { a[i] = alo[i]; a[8 + i] = ahi[i]; }
    v16h bf = *reinterpret_cast<const v16h*>(brow + k + (hi << 4));
    acc = __builtin_amdgcn_wmma_f32_16x16x32_f16(
        false, a, false, bf, (short)0, acc, false, false);
  }

  const float bv = bias[n0 + lh];
  #pragma unroll
  for (int r = 0; r < 8; ++r) {
    int mm = r + (hi << 3);
    out1[(size_t)(m0 + mm) * 128 + (n0 + lh)] = acc[r] + bv;
  }
}

// ---------------------------------------------------------------------------
// 5) FC2 (128 -> 1) + sigmoid.  One thread per batch element.
// ---------------------------------------------------------------------------
__global__ void fc2_kernel(const float* __restrict__ out1,
                           const float* __restrict__ w,
                           const float* __restrict__ b,
                           float* __restrict__ out) {
  int i = threadIdx.x;              // 0..127
  float acc = b[0];
  #pragma unroll 8
  for (int n = 0; n < 128; ++n) acc += out1[i * 128 + n] * w[n];
  out[i] = 1.0f / (1.0f + expf(-acc));
}

// ---------------------------------------------------------------------------
extern "C" void kernel_launch(void* const* d_in, const int* in_sizes, int n_in,
                              void* d_out, int out_size, void* d_ws, size_t ws_size,
                              hipStream_t stream) {
  const int*   inputs    = (const int*)  d_in[0];   // [128,50,9,20]
  const float* emb_table = (const float*)d_in[1];   // [1M,256]
  const float* conv_w    = (const float*)d_in[2];   // [128,256,3,2]
  const float* conv_b    = (const float*)d_in[3];   // [128]
  const float* fc1_w     = (const float*)d_in[4];   // [128,12288]
  const float* fc1_b     = (const float*)d_in[5];   // [128]
  const float* fc2_w     = (const float*)d_in[6];   // [1,128]
  const float* fc2_b     = (const float*)d_in[7];   // [1]
  float* out = (float*)d_out;                       // [128,1]

  char* ws = (char*)d_ws;
  size_t off = 0;
  _Float16* emb    = (_Float16*)(ws + off); off += (size_t)BB * HH * WW * EE * 2;   // 29,491,200
  _Float16* convw  = (_Float16*)(ws + off); off += (size_t)128 * 1536 * 2;          //    393,216
  _Float16* fc1wh  = (_Float16*)(ws + off); off += (size_t)128 * 12288 * 2;         //  3,145,728
  _Float16* pooled = (_Float16*)(ws + off); off += (size_t)128 * 12288 * 2;         //  3,145,728
  float*    out1   = (float*)   (ws + off); off += (size_t)128 * 128 * 4;           //     65,536

  // 1) gather + mean  (HBM-bound, ~50us floor)
  gather_mean_kernel<<<BB * HH * WW, 256, 0, stream>>>(inputs, emb_table, emb);
  // 2) weight repack to f16 WMMA-friendly layouts
  convw_transform<<<(128 * 1536 + 255) / 256, 256, 0, stream>>>(conv_w, convw);
  fc1w_transform<<<(128 * 12288 + 255) / 256, 256, 0, stream>>>(fc1_w, fc1wh);
  // 3) conv as implicit GEMM (WMMA f16, f32 accum) + fused bias/pool/relu
  conv_wmma_kernel<<<3072, 256, 0, stream>>>(emb, convw, conv_b, pooled);
  // 4) FC1 (WMMA)
  fc1_wmma_kernel<<<8, 256, 0, stream>>>(pooled, fc1wh, fc1_b, out1);
  // 5) FC2 + sigmoid
  fc2_kernel<<<1, 128, 0, stream>>>(out1, fc2_w, fc2_b, out);
}
